// WanSelfAttention_pt_32770600468939
// MI455X (gfx1250) — matrix operand order repaired and checked by reading, not compiled, not run on hardware
//
#include <hip/hip_runtime.h>
#include <hip/hip_bf16.h>

// ---------------------------------------------------------------------------
// WanSelfAttention for MI455X (gfx1250, wave32, WMMA + TDM/async staging).
// Pipeline:
//   1) f32->f16 conversion of x and the four weight matrices
//   2) WMMA GEMM  Qraw = x@wq^T+bq (f32), Kraw likewise, V = x@wv^T+bv (f16)
//   3) RMSNorm (full 1536-row) + RoPE  -> qh/kh f16
//   4) flash attention per (head, 128-query block):
//        double-buffered staging: K tile via Tensor Data Mover, V tile via
//        global_load_async_to_lds_b128 -- tile k+1's DMA issued before
//        computing tile k, so the data movers overlap the WMMA work.
//   5) WMMA GEMM  out = attn@wo^T+bo (f32)
// ---------------------------------------------------------------------------

typedef __attribute__((ext_vector_type(8)))  _Float16 v8h;
typedef __attribute__((ext_vector_type(16))) _Float16 v16h;
typedef __attribute__((ext_vector_type(8)))  float    v8f;
typedef __attribute__((ext_vector_type(4)))  unsigned int v4u;
typedef __attribute__((ext_vector_type(4)))  int      v4i;
typedef __attribute__((ext_vector_type(8)))  int      v8i;

#define DIMN   1536
#define HEADSN 12
#define HDN    128
#define SEQN   4096

#if defined(__gfx1250__) && __has_builtin(__builtin_amdgcn_tensor_load_to_lds)
#define HAVE_TDM 1
#else
#define HAVE_TDM 0
#endif

#if defined(__gfx1250__) && __has_builtin(__builtin_amdgcn_global_load_async_to_lds_b128)
#define HAVE_ASYNC 1
#else
#define HAVE_ASYNC 0
#endif

#if defined(__gfx1250__) && __has_builtin(__builtin_amdgcn_s_wait_asynccnt)
#define WAIT_ASYNC() __builtin_amdgcn_s_wait_asynccnt(0)
#else
#define WAIT_ASYNC() asm volatile("s_wait_asynccnt 0x0" ::: "memory")
#endif

__device__ __forceinline__ v8f wmma16(v16h a, v16h b, v8f c) {
  return __builtin_amdgcn_wmma_f32_16x16x32_f16(false, a, false, b,
                                                (short)0, c, false, false);
}

__device__ __forceinline__ v16h cat8(v8h lo, v8h hi) {
  return __builtin_shufflevector(lo, hi, 0,1,2,3,4,5,6,7,8,9,10,11,12,13,14,15);
}

#if HAVE_TDM
// 2D TDM tile load: tile_d0 x tile_d1 elements of 2 bytes, global row stride
// `row_stride` elements, into LDS byte offset `lds_off`. D# per CDNA5 ISA 8.3/8.4.
__device__ __forceinline__ void tdm_load_tile_2d(const void* gaddr,
                                                 unsigned lds_off,
                                                 unsigned tile_d0,
                                                 unsigned tile_d1,
                                                 unsigned row_stride) {
  unsigned long long ga = (unsigned long long)gaddr;
  v4u g0;
  g0[0] = 1u;                                   // count=1, user descriptor
  g0[1] = lds_off;                              // lds_addr (bytes)
  g0[2] = (unsigned)(ga & 0xFFFFFFFFu);         // global_addr[31:0]
  g0[3] = (unsigned)((ga >> 32) & 0x1FFFFFFu) | (2u << 30);  // [56:32] | type=2
  v8i g1;
  g1[0] = (int)(1u << 16);                      // wg_mask=0, data_size=1 (2B)
  g1[1] = (int)((tile_d0 & 0xFFFFu) << 16);     // atomic_barrier=0 | td0[15:0]
  g1[2] = (int)(((tile_d0 >> 16) & 0xFFFFu) |   // tensor_dim0[31:16]
                ((tile_d1 & 0xFFFFu) << 16));   // tensor_dim1[15:0]
  g1[3] = (int)(((tile_d1 >> 16) & 0xFFFFu) |   // tensor_dim1[31:16]
                ((tile_d0 & 0xFFFFu) << 16));   // tile_dim0
  g1[4] = (int)(tile_d1 & 0xFFFFu);             // tile_dim1 | tile_dim2=0
  g1[5] = (int)row_stride;                      // tensor_dim0_stride[31:0]
  g1[6] = 0;                                    // stride0 hi | stride1 lo
  g1[7] = 0;                                    // stride1 hi
  v4i gz = {0, 0, 0, 0};
#if __has_include(<hip/amd_detail/amd_gfx1250_TDM.h>)
  v8i gz8 = {0, 0, 0, 0, 0, 0, 0, 0};
  __builtin_amdgcn_tensor_load_to_lds(g0, g1, gz, gz, gz8, 0);   // 6-arg lane
#else
  __builtin_amdgcn_tensor_load_to_lds(g0, g1, gz, gz, 0);        // 5-arg lane
#endif
}
#endif

// Stage one 32x128 f16 K tile into an LDS buffer (TDM or plain copy).
__device__ __forceinline__ void stage_k_tile(const _Float16* Kb, int head,
                                             int k0, _Float16* KsBuf,
                                             unsigned lds_byte_off,
                                             int skey, int sfb) {
#if HAVE_TDM
  (void)KsBuf; (void)skey; (void)sfb;
  if (threadIdx.x < 32)
    tdm_load_tile_2d(Kb + (size_t)k0 * DIMN + head * HDN, lds_byte_off,
                     /*tile_d0=*/128u, /*tile_d1=*/32u,
                     /*row_stride=*/(unsigned)DIMN);
#else
  (void)lds_byte_off;
  v16h kv = *(const v16h*)(Kb + (size_t)(k0 + skey) * DIMN + head * HDN + sfb);
  *(v16h*)(KsBuf + skey * 128 + sfb) = kv;
#endif
}

// Stage one 32x128 f16 V tile (row-major) into an LDS buffer (async or plain).
__device__ __forceinline__ void stage_v_tile(const _Float16* Vb, int head,
                                             int k0, _Float16* VrBuf,
                                             int skey, int sfb) {
  const _Float16* gsrc = Vb + (size_t)(k0 + skey) * DIMN + head * HDN + sfb;
  _Float16* ldst = VrBuf + skey * 128 + sfb;
#if HAVE_ASYNC
  __builtin_amdgcn_global_load_async_to_lds_b128((v4i*)gsrc, (v4i*)ldst, 0, 0);
  __builtin_amdgcn_global_load_async_to_lds_b128((v4i*)(gsrc + 8),
                                                 (v4i*)(ldst + 8), 0, 0);
#else
  *(v16h*)ldst = *(const v16h*)gsrc;
#endif
}

// Wait for this wave's outstanding staging DMA (TENSORcnt + ASYNCcnt).
__device__ __forceinline__ void wait_stage() {
#if HAVE_TDM
  if (threadIdx.x < 32) __builtin_amdgcn_s_wait_tensorcnt(0);
#endif
#if HAVE_ASYNC
  WAIT_ASYNC();
#endif
}

// ---------------------------------------------------------------------------
__global__ void cvt_f32_f16(const float* __restrict__ in,
                            _Float16* __restrict__ out, int n) {
  int i = blockIdx.x * blockDim.x + threadIdx.x;
  if (i < n) out[i] = (_Float16)in[i];
}

// ---------------------------------------------------------------------------
// C[m,n] = sum_k A[m,k]*W[n,k] + bias[n]
// A: [M,K] f16 row-major, W: [N,K] f16 row-major (weight as given).
// mode 0 -> f32 output, mode 1 -> f16 output.
// block = 256 threads (8 waves), block tile 128(M) x 64(N), wave tile 32x32.
__global__ void __launch_bounds__(256)
gemm_wmma(const _Float16* __restrict__ A, const _Float16* __restrict__ W,
          const float* __restrict__ bias, float* __restrict__ outF,
          _Float16* __restrict__ outH, int M, int N, int K, int mode) {
  const int w    = threadIdx.x >> 5;
  const int lane = threadIdx.x & 31;
  const int hi16 = (lane >> 4) & 1;
  const int ln   = lane & 15;
  const int m0   = blockIdx.y * 128 + (w & 3) * 32;
  const int n0   = blockIdx.x * 64  + (w >> 2) * 32;

  v8f c00 = {}, c01 = {}, c10 = {}, c11 = {};
  const int ca = hi16 ? 16 : 0;   // A-frag K offset per half
  const int cb = hi16 ? 16 : 0;   // B-frag K offset per half

  const _Float16* Ar0 = A + (size_t)(m0 + ln) * K + ca;
  const _Float16* Ar1 = Ar0 + (size_t)16 * K;
  const _Float16* Wr0 = W + (size_t)(n0 + ln) * K + cb;
  const _Float16* Wr1 = Wr0 + (size_t)16 * K;

  for (int kk = 0; kk < K; kk += 32) {
    v16h a0 = cat8(*(const v8h*)(Ar0 + kk), *(const v8h*)(Ar0 + kk + 8));
    v16h a1 = cat8(*(const v8h*)(Ar1 + kk), *(const v8h*)(Ar1 + kk + 8));
    v16h b0 = *(const v16h*)(Wr0 + kk);
    v16h b1 = *(const v16h*)(Wr1 + kk);
    c00 = wmma16(a0, b0, c00);
    c01 = wmma16(a0, b1, c01);
    c10 = wmma16(a1, b0, c10);
    c11 = wmma16(a1, b1, c11);
  }

  const float bv0 = bias[n0 + ln];
  const float bv1 = bias[n0 + 16 + ln];
  const int mb = m0 + hi16 * 8;
  for (int v = 0; v < 8; ++v) {
    const int r0 = mb + v, r1 = r0 + 16;
    float e00 = c00[v] + bv0, e01 = c01[v] + bv1;
    float e10 = c10[v] + bv0, e11 = c11[v] + bv1;
    size_t o00 = (size_t)r0 * N + n0 + ln;
    size_t o01 = o00 + 16;
    size_t o10 = (size_t)r1 * N + n0 + ln;
    size_t o11 = o10 + 16;
    if (mode == 0) {
      outF[o00] = e00; outF[o01] = e01; outF[o10] = e10; outF[o11] = e11;
    } else {
      outH[o00] = (_Float16)e00; outH[o01] = (_Float16)e01;
      outH[o10] = (_Float16)e10; outH[o11] = (_Float16)e11;
    }
  }
}

// ---------------------------------------------------------------------------
// Per-row RMSNorm over DIM then RoPE (F=H=W=16 grid), write f16.
__global__ void __launch_bounds__(256)
rms_rope(const float* __restrict__ raw, const float* __restrict__ g,
         const float* __restrict__ freqs, _Float16* __restrict__ out) {
  __shared__ float red[256];
  const int row = blockIdx.x, tid = threadIdx.x;
  const float* x = raw + (size_t)row * DIMN;

  float ss = 0.f;
  for (int i = tid; i < DIMN; i += 256) { float v = x[i]; ss += v * v; }
  red[tid] = ss;
  __syncthreads();
  for (int s = 128; s > 0; s >>= 1) {
    if (tid < s) red[tid] += red[tid + s];
    __syncthreads();
  }
  const float inv = rsqrtf(red[0] * (1.0f / DIMN) + 1e-6f);

  const int f = row >> 8, h = (row >> 4) & 15, wg = row & 15;
  for (int p = tid; p < DIMN / 2; p += 256) {
    const int head = p >> 6, c = p & 63;
    const int sel = (c < 22) ? f : ((c < 43) ? h : wg);
    const float ang = freqs[sel * 64 + c];
    float sn, cs;
    __sincosf(ang, &sn, &cs);
    const int i0 = head * HDN + 2 * c;
    const float xr = x[i0]     * inv * g[i0];
    const float xi = x[i0 + 1] * inv * g[i0 + 1];
    out[(size_t)row * DIMN + i0]     = (_Float16)(xr * cs - xi * sn);
    out[(size_t)row * DIMN + i0 + 1] = (_Float16)(xr * sn + xi * cs);
  }
}

// ---------------------------------------------------------------------------
// Flash attention. grid = (HEADS, SEQ/128); 8 waves, 16 query rows per wave.
// Key tiles of 32 (= one WMMA K step), double-buffered DMA staging.
// LDS layout (halves): Ks0 @0, Ks1 @4096, Vr0 @8192, Vr1 @12288,
//                      Vt @16384, P [8 waves x 16x32] @20480.   (48 KB)
__global__ void __launch_bounds__(256)
flash_attn(const _Float16* __restrict__ Q, const _Float16* __restrict__ Kb,
           const _Float16* __restrict__ Vb, _Float16* __restrict__ O,
           const int* __restrict__ seq_lens) {
  __shared__ _Float16 lds[24576];
  _Float16* Ks0 = lds;                     // [32 keys][128 feat]
  _Float16* Ks1 = lds + 4096;
  _Float16* Vr0 = lds + 8192;              // [32 keys][128 feat] row-major
  _Float16* Vr1 = lds + 12288;
  _Float16* Vt  = lds + 16384;             // [128 feat][32 keys] transposed
  const int w    = threadIdx.x >> 5;
  const int lane = threadIdx.x & 31;
  _Float16* Pw = lds + 20480 + w * 512;    // [16 q][32 keys] per wave
  const int hi16 = (lane >> 4) & 1;
  const int ln   = lane & 15;
  const int head = blockIdx.x;
  const int q0   = blockIdx.y * 128 + w * 16;
  const int seqlen = seq_lens[0];
  const float scale = 0.08838834764831845f;   // 1/sqrt(128)

  // Q fragments, kept in registers for the whole kernel.
  const _Float16* qrow = Q + (size_t)(q0 + ln) * DIMN + head * HDN;
  const int ca = hi16 ? 16 : 0;
  v16h qf[4];
  for (int kf = 0; kf < 4; ++kf)
    qf[kf] = cat8(*(const v8h*)(qrow + kf * 32 + ca),
                  *(const v8h*)(qrow + kf * 32 + ca + 8));

  v8f acc[8] = {};
  float mrun[8], lrun[8];
  for (int v = 0; v < 8; ++v) { mrun[v] = -3.0e38f; lrun[v] = 0.f; }

  const int skey = threadIdx.x >> 3;        // staging: key 0..31
  const int sfb  = (threadIdx.x & 7) * 16;  // staging: feature base

  // prologue: stage tile 0 into buffer 0
  stage_k_tile(Kb, head, 0, Ks0, 0u, skey, sfb);
  stage_v_tile(Vb, head, 0, Vr0, skey, sfb);

  for (int k0 = 0; k0 < SEQN; k0 += 32) {
    const int buf = (k0 >> 5) & 1;
    _Float16* KsBuf = buf ? Ks1 : Ks0;
    _Float16* VrBuf = buf ? Vr1 : Vr0;

    wait_stage();        // tile k0 DMA complete (this wave's counters)
    __syncthreads();     // staged data visible; prev tile's readers done

    // transpose V current tile into Vt (each thread handles its own chunk)
    {
      v16h vv = *(const v16h*)(VrBuf + skey * 128 + sfb);
      for (int i = 0; i < 16; ++i) Vt[(sfb + i) * 32 + skey] = vv[i];
    }

    // prefetch tile k0+32 into the other buffers (overlaps WMMA below)
    if (k0 + 32 < SEQN) {
      stage_k_tile(Kb, head, k0 + 32, buf ? Ks0 : Ks1,
                   buf ? 0u : 8192u, skey, sfb);
      stage_v_tile(Vb, head, k0 + 32, buf ? Vr0 : Vr1, skey, sfb);
    }
    __syncthreads();     // Vt ready for all waves

    // ---- scores for two 16-key sub-tiles ----
    v8f s0 = {}, s1 = {};
    const int cb = hi16 ? 16 : 0;
    for (int kf = 0; kf < 4; ++kf) {
      v16h b0 = *(const v16h*)(KsBuf + ln * 128 + kf * 32 + cb);
      v16h b1 = *(const v16h*)(KsBuf + (16 + ln) * 128 + kf * 32 + cb);
      s0 = wmma16(qf[kf], b0, s0);
      s1 = wmma16(qf[kf], b1, s1);
    }

    const float madd0 = (k0 + ln      < seqlen) ? 0.f : -1.0e9f;
    const float madd1 = (k0 + 16 + ln < seqlen) ? 0.f : -1.0e9f;

    // ---- online softmax per row (row r = v + hi16*8, across a 16-lane half)
    for (int v = 0; v < 8; ++v) {
      float e0 = s0[v] * scale + madd0;
      float e1 = s1[v] * scale + madd1;
      float mx = fmaxf(e0, e1);
      for (int off = 1; off < 16; off <<= 1)
        mx = fmaxf(mx, __shfl_xor(mx, off, 32));
      float mnew = fmaxf(mrun[v], mx);
      float al = __expf(mrun[v] - mnew);
      float p0 = __expf(e0 - mnew);
      float p1 = __expf(e1 - mnew);
      float rs = p0 + p1;
      for (int off = 1; off < 16; off <<= 1)
        rs += __shfl_xor(rs, off, 32);
      lrun[v] = lrun[v] * al + rs;
      mrun[v] = mnew;
      for (int t = 0; t < 8; ++t) acc[t][v] *= al;
      const int r = v + hi16 * 8;
      Pw[r * 32 + ln]      = (_Float16)p0;
      Pw[r * 32 + 16 + ln] = (_Float16)p1;
    }

    // ---- P as A-fragment (C-layout -> A-layout via LDS) ----
    v16h pf = cat8(*(const v8h*)(Pw + ln * 32 + ca),
                   *(const v8h*)(Pw + ln * 32 + ca + 8));

    // ---- acc += P @ V  (8 feature tiles of 16) ----
    const int cv = hi16 ? 16 : 0;
    for (int t = 0; t < 8; ++t) {
      v16h bvf = *(const v16h*)(Vt + (t * 16 + ln) * 32 + cv);
      acc[t] = wmma16(pf, bvf, acc[t]);
    }
  }

  // ---- normalize and store ----
  for (int v = 0; v < 8; ++v) {
    const int r = v + hi16 * 8;
    const float inv = 1.0f / lrun[v];
    size_t base = (size_t)(q0 + r) * DIMN + head * HDN + ln;
    for (int t = 0; t < 8; ++t)
      O[base + t * 16] = (_Float16)(acc[t][v] * inv);
  }
}

// ---------------------------------------------------------------------------
extern "C" void kernel_launch(void* const* d_in, const int* in_sizes, int n_in,
                              void* d_out, int out_size, void* d_ws,
                              size_t ws_size, hipStream_t stream) {
  const float* x        = (const float*)d_in[0];
  const int*   seq_lens = (const int*)  d_in[1];
  const float* freqs    = (const float*)d_in[3];
  const float* wq = (const float*)d_in[4];
  const float* bq = (const float*)d_in[5];
  const float* wk = (const float*)d_in[6];
  const float* bk = (const float*)d_in[7];
  const float* wv = (const float*)d_in[8];
  const float* bv = (const float*)d_in[9];
  const float* wo = (const float*)d_in[10];
  const float* bo = (const float*)d_in[11];
  const float* gq = (const float*)d_in[12];
  const float* gk = (const float*)d_in[13];

  char* ws = (char*)d_ws;
  size_t off = 0;
  auto take = [&](size_t bytes) -> void* {
    void* p = ws + off;
    off += (bytes + 255) & ~(size_t)255;
    return p;
  };

  const size_t SD2 = (size_t)SEQN * DIMN * 2;   // f16 [S,DIM]
  const size_t SD4 = (size_t)SEQN * DIMN * 4;   // f32 [S,DIM]
  const size_t WW2 = (size_t)DIMN * DIMN * 2;   // f16 [DIM,DIM]

  _Float16* xh   = (_Float16*)take(SD2);
  _Float16* wqh  = (_Float16*)take(WW2);
  _Float16* wkh  = (_Float16*)take(WW2);
  _Float16* wvh  = (_Float16*)take(WW2);
  _Float16* woh  = (_Float16*)take(WW2);
  float*    qraw = (float*)   take(SD4);
  float*    kraw = (float*)   take(SD4);
  _Float16* qh   = (_Float16*)take(SD2);
  _Float16* kh   = (_Float16*)take(SD2);
  _Float16* vh   = (_Float16*)take(SD2);
  _Float16* attn = (_Float16*)take(SD2);

  // 1) precision conversion
  {
    int n = SEQN * DIMN;
    cvt_f32_f16<<<(n + 255) / 256, 256, 0, stream>>>(x, xh, n);
    n = DIMN * DIMN;
    cvt_f32_f16<<<(n + 255) / 256, 256, 0, stream>>>(wq, wqh, n);
    cvt_f32_f16<<<(n + 255) / 256, 256, 0, stream>>>(wk, wkh, n);
    cvt_f32_f16<<<(n + 255) / 256, 256, 0, stream>>>(wv, wvh, n);
    cvt_f32_f16<<<(n + 255) / 256, 256, 0, stream>>>(wo, woh, n);
  }

  // 2) projections
  dim3 gg(DIMN / 64, SEQN / 128);
  gemm_wmma<<<gg, 256, 0, stream>>>(xh, wqh, bq, qraw, nullptr,
                                    SEQN, DIMN, DIMN, 0);
  gemm_wmma<<<gg, 256, 0, stream>>>(xh, wkh, bk, kraw, nullptr,
                                    SEQN, DIMN, DIMN, 0);
  gemm_wmma<<<gg, 256, 0, stream>>>(xh, wvh, bv, nullptr, vh,
                                    SEQN, DIMN, DIMN, 1);

  // 3) RMSNorm + RoPE
  rms_rope<<<SEQN, 256, 0, stream>>>(qraw, gq, freqs, qh);
  rms_rope<<<SEQN, 256, 0, stream>>>(kraw, gk, freqs, kh);

  // 4) attention
  dim3 ga(HEADSN, SEQN / 128);
  flash_attn<<<ga, 256, 0, stream>>>(qh, kh, vh, attn, seq_lens);

  // 5) output projection
  gemm_wmma<<<gg, 256, 0, stream>>>(attn, woh, bo, (float*)d_out, nullptr,
                                    SEQN, DIMN, DIMN, 0);

  (void)in_sizes; (void)n_in; (void)out_size; (void)ws_size;
}
